// MultiHeadAttention_48086453846410
// MI455X (gfx1250) — matrix operand order repaired and checked by reading, not compiled, not run on hardware
//
#include <hip/hip_runtime.h>
#include <math.h>

// ---------------------------------------------------------------------------
// CDNA5 / gfx1250 types for V_WMMA_F32_16X16X32_BF16  (wave32)
// ---------------------------------------------------------------------------
typedef __attribute__((ext_vector_type(16))) __bf16 v16bf;
typedef __attribute__((ext_vector_type(8)))  float  v8f;
typedef __attribute__((ext_vector_type(4)))  int    v4i;

union FragB { uint4 q[2]; v16bf v; };        // 8 VGPRs of bf16 operand data

// ---- optional gfx1250 async global->LDS path (ASYNCcnt) -------------------
#if defined(__has_builtin)
#if __has_builtin(__builtin_amdgcn_global_load_async_to_lds_b128) && \
    __has_builtin(__builtin_amdgcn_s_wait_asynccnt)
#define USE_ASYNC_COPY 1
#endif
#endif
#ifndef USE_ASYNC_COPY
#define USE_ASYNC_COPY 0
#endif

#if USE_ASYNC_COPY
__device__ __forceinline__ void async_cp16(const void* g, void* l) {
  // per-lane: LDS[l] = MEM[g], 16 bytes, tracked by ASYNCcnt
  __builtin_amdgcn_global_load_async_to_lds_b128(
      (__attribute__((address_space(1))) v4i*)g,
      (__attribute__((address_space(3))) v4i*)l, 0, 0);
}
#define ASYNC_WAIT() __builtin_amdgcn_s_wait_asynccnt(0)
#else
__device__ __forceinline__ void async_cp16(const void* g, void* l) {
  *(uint4*)l = *(const uint4*)g;
}
#define ASYNC_WAIT() ((void)0)
#endif

__device__ __forceinline__ unsigned short f2bf(float f) {
  unsigned int u = __float_as_uint(f);
  u += 0x7FFFu + ((u >> 16) & 1u);           // round-to-nearest-even
  return (unsigned short)(u >> 16);
}

__device__ __forceinline__ v8f wmma_bf16(const FragB& a, const FragB& b, v8f c) {
  // (neg_a, A, neg_b, B, c_mod, C, reuse_a, reuse_b)
  return __builtin_amdgcn_wmma_f32_16x16x32_bf16(false, a.v, false, b.v,
                                                 (short)0, c, false, false);
}

// ---------------------------------------------------------------------------
// f32 -> bf16 bulk convert
// ---------------------------------------------------------------------------
__global__ __launch_bounds__(256)
void cvt_f32_bf16(const float* __restrict__ src, unsigned short* __restrict__ dst, int n) {
  int i = (blockIdx.x * 256 + threadIdx.x) * 4;
  if (i + 3 < n) {
    float4 f = *(const float4*)(src + i);
    uint2 pk;
    pk.x = (unsigned)f2bf(f.x) | ((unsigned)f2bf(f.y) << 16);
    pk.y = (unsigned)f2bf(f.z) | ((unsigned)f2bf(f.w) << 16);
    *(uint2*)(dst + i) = pk;
  }
}

// ---------------------------------------------------------------------------
// C[b] = (A[b] (MxK) @ W^T (NxK) + bias) * scale       bf16 in, f32 acc
// Block tile 128x128, K-step 32, 8 waves in 2(M) x 4(N); wave tile 64x32.
// OMODE: 0 = bf16 row-major (m,n);  1 = f32 row-major (m,n);
//        2 = bf16 attention-transposed for V: vT[n&511][m*8 + (n>>9)]
//            (i.e. vT[d][l] with l = s*8+h, n = h*512+d, m = s; N must be 4096)
// ---------------------------------------------------------------------------
template <int OMODE>
__global__ __launch_bounds__(256)
void gemm_bias(const unsigned short* __restrict__ A,
               const unsigned short* __restrict__ W,
               const float* __restrict__ bias,
               void* __restrict__ Cout,
               int M, int N, int K,
               long long strideA, long long strideC, float scale)
{
  constexpr int BM = 128, BN = 128, BK = 32, LDT = BK + 8;   // +16B row pad
  __shared__ __align__(16) unsigned short As[BM * LDT];
  __shared__ __align__(16) unsigned short Bs[BN * LDT];

  const int tid  = threadIdx.x;
  const int b    = blockIdx.z;
  const int m0   = blockIdx.y * BM;
  const int n0   = blockIdx.x * BN;
  const int lane = tid & 31, wave = tid >> 5;
  const int hl   = lane >> 4, lrow = lane & 15;
  const int mwave = (wave >> 2) * 64;          // 2 waves in M
  const int nwave = (wave & 3) * 32;           // 4 waves in N

  const unsigned short* Ab = A + (long long)b * strideA;

  v8f vz = {0.f, 0.f, 0.f, 0.f, 0.f, 0.f, 0.f, 0.f};
  v8f acc[4][2];
#pragma unroll
  for (int i = 0; i < 4; ++i)
#pragma unroll
    for (int j = 0; j < 2; ++j) acc[i][j] = vz;

  for (int kb = 0; kb < K; kb += BK) {
    // cooperative tile loads: 128x32 bf16 each = 512 x 16B, 2 per thread
#pragma unroll
    for (int i = 0; i < 2; ++i) {
      int idx = tid + i * 256;
      int r = idx >> 2, c = (idx & 3) * 8;
      async_cp16(&Ab[(long long)(m0 + r) * K + kb + c], &As[r * LDT + c]);
      async_cp16(&W [(long long)(n0 + r) * K + kb + c], &Bs[r * LDT + c]);
    }
    ASYNC_WAIT();
    __syncthreads();

    // prefetch next K tile into caches
    if (kb + BK < K) {
      int r = tid >> 2, c = (tid & 3) * 8;
      __builtin_prefetch(&Ab[(long long)(m0 + r) * K + kb + BK + c], 0, 1);
      __builtin_prefetch(&W [(long long)(n0 + r) * K + kb + BK + c], 0, 1);
    }

    FragB af[4], bf[2];
#pragma unroll
    for (int fm = 0; fm < 4; ++fm) {           // A: lane<16 K{0..7,16..23}
      int r = mwave + fm * 16 + lrow;
      af[fm].q[0] = *(const uint4*)&As[r * LDT + hl * 8];
      af[fm].q[1] = *(const uint4*)&As[r * LDT + 16 + hl * 8];
    }
#pragma unroll
    for (int fn = 0; fn < 2; ++fn) {           // B: lane<16 K{0..7,16..23}
      int r = nwave + fn * 16 + lrow;
      bf[fn].q[0] = *(const uint4*)&Bs[r * LDT + hl * 8];
      bf[fn].q[1] = *(const uint4*)&Bs[r * LDT + 16 + hl * 8];
    }
#pragma unroll
    for (int fm = 0; fm < 4; ++fm)
#pragma unroll
      for (int fn = 0; fn < 2; ++fn)
        acc[fm][fn] = wmma_bf16(af[fm], bf[fn], acc[fm][fn]);
    __syncthreads();
  }

  // epilogue: C layout = VGPR j -> row j (+8 for upper half-wave), col = lrow
#pragma unroll
  for (int fm = 0; fm < 4; ++fm) {
#pragma unroll
    for (int fn = 0; fn < 2; ++fn) {
      int n = n0 + nwave + fn * 16 + lrow;
      float bv = bias[n];
#pragma unroll
      for (int j = 0; j < 8; ++j) {
        int m = m0 + mwave + fm * 16 + hl * 8 + j;
        float val = (acc[fm][fn][j] + bv) * scale;
        if (OMODE == 1) {
          ((float*)Cout)[(long long)b * strideC + (long long)m * N + n] = val;
        } else if (OMODE == 0) {
          ((unsigned short*)Cout)[(long long)b * strideC + (long long)m * N + n] = f2bf(val);
        } else {                       // OMODE==2: vT[d][l], d=n&511, l=m*8+(n>>9)
          long long idx = (long long)b * strideC +
                          (long long)(n & 511) * 4096 + (long long)m * 8 + (n >> 9);
          ((unsigned short*)Cout)[idx] = f2bf(val);
        }
      }
    }
  }
}

// ---------------------------------------------------------------------------
// Flash attention: per block = 32 query rows (of L=4096), full Dh=512.
// Key-block 128: waves own 16-key slices for S (full-depth via 64-wide
// d-chunks), online softmax in f32, O += P@V with waves owning 64 D-cols.
// qp already carries 1/sqrt(512).  V arrives PRE-TRANSPOSED: VpT[b][d][l].
// ---------------------------------------------------------------------------
__global__ __launch_bounds__(256)
void flash_attn(const unsigned short* __restrict__ Qp,
                const unsigned short* __restrict__ Kp,
                const unsigned short* __restrict__ VpT,
                unsigned short* __restrict__ O)
{
  constexpr int Dh = 512, L = 4096, BR = 32, BC = 128;
  constexpr int LQ  = Dh + 8;    // Q row stride (bf16 elems)
  constexpr int LKT = 64 + 8;    // K d-chunk row stride
  constexpr int LVT = 32 + 8;    // V^T row stride
  constexpr int LS  = BC + 8;    // S / P row stride

  __shared__ __align__(16) unsigned short Qs[BR * LQ];     // 33.3 KB
  __shared__ __align__(16) unsigned short Kt[BC * LKT];    // 18.4 KB
  __shared__ __align__(16) unsigned short Vt[Dh * LVT];    // 41.0 KB
  __shared__ __align__(16) float          Sred[BR * LS];   // 17.4 KB
  __shared__ __align__(16) unsigned short P[BR * LS];      //  8.7 KB
  __shared__ float corrS[BR];
  __shared__ float lFin[BR];

  const int tid  = threadIdx.x;
  const int b    = blockIdx.y;
  const int q0   = blockIdx.x * BR;
  const int lane = tid & 31, wave = tid >> 5;
  const int hl   = lane >> 4, lrow = lane & 15;

  const unsigned short* Qb  = Qp  + ((long long)b * L + q0) * Dh;
  const unsigned short* Kb  = Kp  + (long long)b * L * Dh;
  const unsigned short* VTb = VpT + (long long)b * (long long)Dh * L;

  // load Q block 32x512 (2048 x 16B, 8 per thread)
#pragma unroll
  for (int i = 0; i < 8; ++i) {
    int idx = tid + i * 256;
    int r = idx >> 6, c = (idx & 63) * 8;
    async_cp16(&Qb[(long long)r * Dh + c], &Qs[r * LQ + c]);
  }

  v8f vz = {0.f, 0.f, 0.f, 0.f, 0.f, 0.f, 0.f, 0.f};
  v8f oacc[2][4];
#pragma unroll
  for (int i = 0; i < 2; ++i)
#pragma unroll
    for (int j = 0; j < 4; ++j) oacc[i][j] = vz;

  float mRow = -INFINITY, lRow = 0.f;          // live only in threads 0..31

  for (int kb = 0; kb < L; kb += BC) {
    // ---------------- S = Q @ K^T  (each wave: 16 keys, full depth) --------
    v8f sacc[2]; sacc[0] = vz; sacc[1] = vz;
    for (int dc = 0; dc < Dh; dc += 64) {
      __syncthreads();                          // Kt WAR (also covers Q load)
#pragma unroll
      for (int i = 0; i < 4; ++i) {             // 128 keys x 64 d = 1024 x 16B
        int idx = tid + i * 256;
        int r = idx >> 3, c = (idx & 7) * 8;
        async_cp16(&Kb[(long long)(kb + r) * Dh + dc + c], &Kt[r * LKT + c]);
      }
      ASYNC_WAIT();
      __syncthreads();
#pragma unroll
      for (int ks = 0; ks < 2; ++ks) {
        FragB bfr;
        int rB = wave * 16 + lrow;              // B: key column, d {0..7,16..23}
        bfr.q[0] = *(const uint4*)&Kt[rB * LKT + ks * 32 + hl * 8];
        bfr.q[1] = *(const uint4*)&Kt[rB * LKT + ks * 32 + 16 + hl * 8];
#pragma unroll
        for (int fm = 0; fm < 2; ++fm) {
          FragB afr;
          int rA = fm * 16 + lrow;
          afr.q[0] = *(const uint4*)&Qs[rA * LQ + dc + ks * 32 + hl * 8];
          afr.q[1] = *(const uint4*)&Qs[rA * LQ + dc + ks * 32 + 16 + hl * 8];
          sacc[fm] = wmma_bf16(afr, bfr, sacc[fm]);
        }
      }
    }
    // publish S slice (f32)
#pragma unroll
    for (int fm = 0; fm < 2; ++fm)
#pragma unroll
      for (int j = 0; j < 8; ++j)
        Sred[(fm * 16 + hl * 8 + j) * LS + wave * 16 + lrow] = sacc[fm][j];
    __syncthreads();

    // ---------------- online softmax (thread t owns row t, float4) ---------
    if (tid < BR) {
      const float4* Srow = (const float4*)&Sred[tid * LS];
      float mx = mRow;
#pragma unroll
      for (int j4 = 0; j4 < BC / 4; ++j4) {
        float4 s = Srow[j4];
        mx = fmaxf(mx, fmaxf(fmaxf(s.x, s.y), fmaxf(s.z, s.w)));
      }
      float cr  = __expf(mRow - mx);
      float sum = lRow * cr;
      uint2* Prow = (uint2*)&P[tid * LS];
#pragma unroll
      for (int j4 = 0; j4 < BC / 4; ++j4) {
        float4 s = Srow[j4];
        float p0 = __expf(s.x - mx), p1 = __expf(s.y - mx);
        float p2 = __expf(s.z - mx), p3 = __expf(s.w - mx);
        sum += (p0 + p1) + (p2 + p3);
        uint2 pk;
        pk.x = (unsigned)f2bf(p0) | ((unsigned)f2bf(p1) << 16);
        pk.y = (unsigned)f2bf(p2) | ((unsigned)f2bf(p3) << 16);
        Prow[j4] = pk;
      }
      mRow = mx; lRow = sum;
      corrS[tid] = cr;
    }
    __syncthreads();

    // ---------------- rescale O accumulators -------------------------------
    float cv[2][8];
#pragma unroll
    for (int fm = 0; fm < 2; ++fm)
#pragma unroll
      for (int j = 0; j < 8; ++j)
        cv[fm][j] = corrS[fm * 16 + hl * 8 + j];
#pragma unroll
    for (int fm = 0; fm < 2; ++fm)
#pragma unroll
      for (int fn = 0; fn < 4; ++fn)
#pragma unroll
        for (int j = 0; j < 8; ++j)
          oacc[fm][fn][j] *= cv[fm][j];

    // ---------------- O += P @ V  (wave owns 64 D-columns) -----------------
    for (int kc = 0; kc < BC; kc += 32) {
      __syncthreads();                          // Vt WAR
#pragma unroll
      for (int i = 0; i < 8; ++i) {             // V^T: 512 d-rows x 32 keys
        int idx = tid + i * 256;
        int r = idx >> 2, c = (idx & 3) * 8;    // r = d, c = key within chunk
        async_cp16(&VTb[(long long)r * L + kb + kc + c], &Vt[r * LVT + c]);
      }
      ASYNC_WAIT();
      __syncthreads();
      FragB pf[2];
#pragma unroll
      for (int fm = 0; fm < 2; ++fm) {
        int rA = fm * 16 + lrow;
        pf[fm].q[0] = *(const uint4*)&P[rA * LS + kc + hl * 8];
        pf[fm].q[1] = *(const uint4*)&P[rA * LS + kc + 16 + hl * 8];
      }
#pragma unroll
      for (int fn = 0; fn < 4; ++fn) {
        FragB vf;
        int rB = wave * 64 + fn * 16 + lrow;    // B: D column, keys {0..7,16..23}
        vf.q[0] = *(const uint4*)&Vt[rB * LVT + hl * 8];
        vf.q[1] = *(const uint4*)&Vt[rB * LVT + 16 + hl * 8];
#pragma unroll
        for (int fm = 0; fm < 2; ++fm)
          oacc[fm][fn] = wmma_bf16(pf[fm], vf, oacc[fm][fn]);
      }
    }
  }

  // ---------------- finalize: O /= l, store bf16 ---------------------------
  if (tid < BR) lFin[tid] = 1.0f / lRow;
  __syncthreads();
  float lv[2][8];
#pragma unroll
  for (int fm = 0; fm < 2; ++fm)
#pragma unroll
    for (int j = 0; j < 8; ++j)
      lv[fm][j] = lFin[fm * 16 + hl * 8 + j];

  unsigned short* Ob = O + ((long long)b * L + q0) * Dh;
#pragma unroll
  for (int fm = 0; fm < 2; ++fm)
#pragma unroll
    for (int fn = 0; fn < 4; ++fn)
#pragma unroll
      for (int j = 0; j < 8; ++j) {
        int m = fm * 16 + hl * 8 + j;
        int n = wave * 64 + fn * 16 + lrow;
        Ob[(long long)m * Dh + n] = f2bf(oacc[fm][fn][j] * lv[fm][j]);
      }
}

// ---------------------------------------------------------------------------
// host launcher
// ---------------------------------------------------------------------------
extern "C" void kernel_launch(void* const* d_in, const int* in_sizes, int n_in,
                              void* d_out, int out_size, void* d_ws, size_t ws_size,
                              hipStream_t stream) {
  (void)in_sizes; (void)n_in; (void)out_size; (void)ws_size;
  const float* q     = (const float*)d_in[0];
  const float* k     = (const float*)d_in[1];
  const float* v     = (const float*)d_in[2];
  const float* wq_w  = (const float*)d_in[3];
  const float* wq_b  = (const float*)d_in[4];
  const float* wk_w  = (const float*)d_in[5];
  const float* wk_b  = (const float*)d_in[6];
  const float* wv_w  = (const float*)d_in[7];
  const float* wv_b  = (const float*)d_in[8];
  const float* out_w = (const float*)d_in[9];
  const float* out_b = (const float*)d_in[10];
  float* out = (float*)d_out;

  const size_t MB = 1024 * 1024;
  char* ws = (char*)d_ws;
  unsigned short* q_bf  = (unsigned short*)(ws + 0 * MB);
  unsigned short* k_bf  = (unsigned short*)(ws + 2 * MB);
  unsigned short* v_bf  = (unsigned short*)(ws + 4 * MB);
  unsigned short* wk_bf = (unsigned short*)(ws + 6 * MB);
  unsigned short* wq_bf = (unsigned short*)(ws + 10 * MB);
  unsigned short* wv_bf = (unsigned short*)(ws + 14 * MB);
  unsigned short* o_bf  = (unsigned short*)(ws + 0 * MB);   // aliases dead bufs
  unsigned short* wo_bf = (unsigned short*)(ws + 18 * MB);
  unsigned short* qp    = (unsigned short*)(ws + 22 * MB);
  unsigned short* kp    = (unsigned short*)(ws + 38 * MB);
  unsigned short* vpT   = (unsigned short*)(ws + 54 * MB);  // [b][d][l], 70 MB total

  const int nAct = 4 * 512 * 512;    // 1,048,576
  const int nW   = 4096 * 512;       // 2,097,152
  cvt_f32_bf16<<<nAct / 4 / 256, 256, 0, stream>>>(q, q_bf, nAct);
  cvt_f32_bf16<<<nAct / 4 / 256, 256, 0, stream>>>(k, k_bf, nAct);
  cvt_f32_bf16<<<nAct / 4 / 256, 256, 0, stream>>>(v, v_bf, nAct);
  cvt_f32_bf16<<<nW / 4 / 256, 256, 0, stream>>>(wk_w, wk_bf, nW);
  cvt_f32_bf16<<<nW / 4 / 256, 256, 0, stream>>>(wq_w, wq_bf, nW);
  cvt_f32_bf16<<<nW / 4 / 256, 256, 0, stream>>>(wv_w, wv_bf, nW);
  cvt_f32_bf16<<<nW / 4 / 256, 256, 0, stream>>>(out_w, wo_bf, nW);

  // projections. NOTE reference swap: qp uses wk_*, kp uses wq_*.
  // 1/sqrt(512) folded into qp.  vp written transposed for the flash kernel.
  dim3 gP(4096 / 128, 512 / 128, 4);
  gemm_bias<0><<<gP, 256, 0, stream>>>(q_bf, wk_bf, wk_b, (void*)qp,
      512, 4096, 512, 512LL * 512, 512LL * 4096, 0.044194173824159216f);
  gemm_bias<0><<<gP, 256, 0, stream>>>(k_bf, wq_bf, wq_b, (void*)kp,
      512, 4096, 512, 512LL * 512, 512LL * 4096, 1.0f);
  gemm_bias<2><<<gP, 256, 0, stream>>>(v_bf, wv_bf, wv_b, (void*)vpT,
      512, 4096, 512, 512LL * 512, 512LL * 4096, 1.0f);

  // flash attention over L=4096, Dh=512 per batch
  flash_attn<<<dim3(4096 / 32, 4), 256, 0, stream>>>(qp, kp, vpT, o_bf);

  // output projection -> f32 d_out
  dim3 gO(512 / 128, 512 / 128, 4);
  gemm_bias<1><<<gO, 256, 0, stream>>>(o_bf, wo_bf, out_b, (void*)out,
      512, 512, 4096, 512LL * 4096, 512LL * 512, 1.0f);
}
